// ENSEMBLE_MODEL_36318243455688
// MI455X (gfx1250) — matrix operand order repaired and checked by reading, not compiled, not run on hardware
//
#include <hip/hip_runtime.h>
#include <math.h>

typedef __bf16  v16bf __attribute__((ext_vector_type(16)));
typedef float   v8f   __attribute__((ext_vector_type(8)));

#define B_ 2
#define N_ 768
#define D_ 64
#define TEMP_INV 0.5f
#define BN_EPS_ 1e-5f
#define SELU_SCALE 1.0507009873554805f
#define SELU_ALPHA 1.6732632423543772f

// 8x gfx1250 hardware tanh (TRANS32), in place. Each tanh is an independent
// instruction covering its predecessor's TRANS hazard; one trailing v_nop
// covers the last (ISA 7.4: 1 independent op or V_NOP before result use).
__device__ __forceinline__ void htanh8_f32(float* v) {
    asm volatile(
        "v_tanh_f32 %0, %0\n\t"
        "v_tanh_f32 %1, %1\n\t"
        "v_tanh_f32 %2, %2\n\t"
        "v_tanh_f32 %3, %3\n\t"
        "v_tanh_f32 %4, %4\n\t"
        "v_tanh_f32 %5, %5\n\t"
        "v_tanh_f32 %6, %6\n\t"
        "v_tanh_f32 %7, %7\n\t"
        "v_nop"
        : "+v"(v[0]), "+v"(v[1]), "+v"(v[2]), "+v"(v[3]),
          "+v"(v[4]), "+v"(v[5]), "+v"(v[6]), "+v"(v[7]));
}

__global__ __launch_bounds__(256)
void ensemble_attn_kernel(const float* __restrict__ x,
                          const float* __restrict__ W_att,
                          const float* __restrict__ b_att,
                          const float* __restrict__ a_w,
                          const float* __restrict__ W_p,
                          const float* __restrict__ b_p,
                          const float* __restrict__ W_q,
                          const float* __restrict__ b_q,
                          const float* __restrict__ gamma,
                          const float* __restrict__ beta,
                          const float* __restrict__ r_mean,
                          const float* __restrict__ r_var,
                          float* __restrict__ out)
{
    __shared__ float s_xi[D_];      // x[b,i,:]
    __shared__ float s_att[N_];     // scores -> attention weights
    __shared__ float s_red[256];    // reduction scratch
    __shared__ float s_agg[256];    // aggregation partials

    const int tid  = threadIdx.x;
    const int wave = tid >> 5;
    const int lane = tid & 31;
    const int g    = lane >> 4;   // lane half (ISA fragment layout)
    const int lrow = lane & 15;

    const int bi = blockIdx.x;
    const int b  = bi / N_;
    const int i  = bi - b * N_;

    if (tid < D_) s_xi[tid] = x[(b * N_ + i) * D_ + tid];
    __syncthreads();

    // ---- Hoist B fragments: W_att[K, o], 4 o-tiles x 2 k-steps (block-invariant)
    // B 32x16 bf16 layout: column N = lane%16, K = 16*(e/8) + 8*(lane/16) + e%8 (+ 32*ks)
    v16bf bfr[8];
#pragma unroll
    for (int ot = 0; ot < 4; ++ot)
#pragma unroll
        for (int ks = 0; ks < 2; ++ks)
#pragma unroll
            for (int e = 0; e < 16; ++e) {
                int K = ((e >> 3) << 4) + (g << 3) + (e & 7) + ks * 32;
                bfr[ot * 2 + ks][e] = (__bf16)W_att[K * D_ + ot * 16 + lrow];
            }

    float ba4[4], aw4[4];
#pragma unroll
    for (int ot = 0; ot < 4; ++ot) {
        ba4[ot] = b_att[ot * 16 + lrow];
        aw4[ot] = a_w [ot * 16 + lrow];
    }

    // ---- Score GEMM: per wave, 16-row j-tiles; A[j,d] = x[j,d]*x_i[d] (bf16)
    for (int jt = wave; jt < N_ / 16; jt += 8) {
        const int j0 = jt * 16;
        const float* xrow = x + (size_t)(b * N_ + j0 + lrow) * D_;

        // A 16x32 bf16 layout: M = lane%16; K = 16*(v/4) + 8*g + 2*(v%4) + slot
        v16bf afr[2];
#pragma unroll
        for (int ks = 0; ks < 2; ++ks)
#pragma unroll
            for (int e = 0; e < 16; ++e) {
                int v = e >> 1, slot = e & 1;
                int K = ((v >> 2) << 4) + (g << 3) + ((v & 3) << 1) + slot + ks * 32;
                afr[ks][e] = (__bf16)(xrow[K] * s_xi[K]);
            }

        float t[8] = {0.f,0.f,0.f,0.f,0.f,0.f,0.f,0.f};
#pragma unroll
        for (int ot = 0; ot < 4; ++ot) {
            v8f c = {};
            c = __builtin_amdgcn_wmma_f32_16x16x32_bf16(
                    false, afr[0], false, bfr[ot * 2 + 0], (short)0, c, false, false);
            c = __builtin_amdgcn_wmma_f32_16x16x32_bf16(
                    false, afr[1], false, bfr[ot * 2 + 1], (short)0, c, false, false);
            // lane's column o = ot*16 + lrow; rows M = r + 8*g
            float h[8];
#pragma unroll
            for (int r = 0; r < 8; ++r) h[r] = c[r] + ba4[ot];
            htanh8_f32(h);
#pragma unroll
            for (int r = 0; r < 8; ++r) t[r] = fmaf(h[r], aw4[ot], t[r]);
        }
        // reduce the o-partials across the 16 lanes of each half-wave
#pragma unroll
        for (int off = 1; off < 16; off <<= 1)
#pragma unroll
            for (int r = 0; r < 8; ++r)
                t[r] += __shfl_xor(t[r], off, 32);
        if (lrow == 0)
#pragma unroll
            for (int r = 0; r < 8; ++r)
                s_att[j0 + g * 8 + r] = t[r] * TEMP_INV;
    }
    __syncthreads();

    // ---- Softmax over j (768 entries)
    float m = -1e30f;
    for (int j = tid; j < N_; j += 256) m = fmaxf(m, s_att[j]);
    s_red[tid] = m; __syncthreads();
    for (int off = 128; off > 0; off >>= 1) {
        if (tid < off) s_red[tid] = fmaxf(s_red[tid], s_red[tid + off]);
        __syncthreads();
    }
    const float mx = s_red[0];
    __syncthreads();

    float sum = 0.f;
    for (int j = tid; j < N_; j += 256) {
        float e = __expf(s_att[j] - mx);   // args <= 0 after max-subtract: fast path safe
        s_att[j] = e;
        sum += e;
    }
    s_red[tid] = sum; __syncthreads();
    for (int off = 128; off > 0; off >>= 1) {
        if (tid < off) s_red[tid] += s_red[tid + off];
        __syncthreads();
    }
    const float inv_sum = 1.0f / s_red[0];

    // ---- agg[d] = sum_j att[j] * x[b,j,d]
    const int d = tid & 63, slice = tid >> 6;
    float acc = 0.f;
    for (int j = slice * (N_ / 4); j < (slice + 1) * (N_ / 4); ++j)
        acc += s_att[j] * x[(size_t)(b * N_ + j) * D_ + d];
    s_agg[tid] = acc;
    __syncthreads();
    if (tid < D_)
        s_agg[tid] = (s_agg[tid] + s_agg[tid + 64] + s_agg[tid + 128] + s_agg[tid + 192]) * inv_sum;
    __syncthreads();

    // ---- y = agg@W_p + x_i@W_q + biases ; BN(eval) ; SELU
    if (tid < D_) {
        const int o = tid;
        float y = b_p[o] + b_q[o];
        for (int dd = 0; dd < D_; ++dd)
            y += s_agg[dd] * W_p[dd * D_ + o] + s_xi[dd] * W_q[dd * D_ + o];
        const float inv_std = rsqrtf(r_var[o] + BN_EPS_);
        y = (y - r_mean[o]) * gamma[o] * inv_std + beta[o];
        y = (y > 0.f) ? SELU_SCALE * y : SELU_SCALE * SELU_ALPHA * (expf(y) - 1.f);
        out[(size_t)(b * N_ + i) * D_ + o] = y;
    }
}

extern "C" void kernel_launch(void* const* d_in, const int* in_sizes, int n_in,
                              void* d_out, int out_size, void* d_ws, size_t ws_size,
                              hipStream_t stream) {
    (void)in_sizes; (void)n_in; (void)out_size; (void)d_ws; (void)ws_size;
    const float* x      = (const float*)d_in[0];
    const float* W_att  = (const float*)d_in[1];
    const float* b_att  = (const float*)d_in[2];
    const float* a_w    = (const float*)d_in[3];
    const float* W_p    = (const float*)d_in[4];
    const float* b_p    = (const float*)d_in[5];
    const float* W_q    = (const float*)d_in[6];
    const float* b_q    = (const float*)d_in[7];
    const float* gamma  = (const float*)d_in[8];
    const float* beta   = (const float*)d_in[9];
    const float* r_mean = (const float*)d_in[10];
    const float* r_var  = (const float*)d_in[11];
    float* out = (float*)d_out;

    dim3 grid(B_ * N_);
    dim3 block(256);
    ensemble_attn_kernel<<<grid, block, 0, stream>>>(
        x, W_att, b_att, a_w, W_p, b_p, W_q, b_q,
        gamma, beta, r_mean, r_var, out);
}
